// MemoryN2N_17755394801765
// MI455X (gfx1250) — matrix operand order repaired and checked by reading, not compiled, not run on hardware
//
#include <hip/hip_runtime.h>
#include <hip/hip_bf16.h>
#include <math.h>

typedef __bf16 bf16_t;
typedef __attribute__((ext_vector_type(16))) __bf16 v16bf;
typedef __attribute__((ext_vector_type(8)))  __bf16 v8bf;
typedef __attribute__((ext_vector_type(8)))  float   v8f;

#define N_TOK   32768   // b*h*w
#define C_DIM   256
#define K_CODE  2048
#define CP      260     // c+4
#define P1      288     // c+4 padded to multiple of 32 (WMMA K granularity)
#define NT_P1   18      // P1/16 output tiles
#define NT_C    16      // C_DIM/16 output tiles
#define HW      4096    // h*w
#define EMA_RATE 0.999f
#define EPS_NORM 1e-12f
#define EPS_CNT  1e-6f

// ---------------- reductions ----------------
__device__ __forceinline__ float wave_reduce_sum(float v) {
#pragma unroll
    for (int o = 16; o; o >>= 1) v += __shfl_xor(v, o, 32);
    return v;
}

__device__ __forceinline__ float block_reduce_sum(float v, float* sred) {
    const int lane = threadIdx.x & 31, wid = threadIdx.x >> 5;
    v = wave_reduce_sum(v);
    if (lane == 0) sred[wid] = v;
    __syncthreads();
    const int nw = (blockDim.x + 31) >> 5;
    if (wid == 0) {
        float t = (lane < nw) ? sred[lane] : 0.f;
        t = wave_reduce_sum(t);
        if (lane == 0) sred[0] = t;
    }
    __syncthreads();
    float tot = sred[0];
    __syncthreads();
    return tot;
}

// ---------------- WMMA fragment loaders ----------------
// A (16x32 bf16): lane L -> row tile_m+(L&15); two contiguous 8-elem runs at
// K = 16*(L>>4) and 16*(L>>4)+8.  B (32x16 bf16): lane L -> column (L&15); one
// contiguous 16-elem run at K = 16*(L>>4); B stored transposed (row = out col).
__device__ __forceinline__ v16bf load_a_frag(const bf16_t* base, int stride,
                                             int tile_m, int k0, int lane) {
    const int m = lane & 15, half = lane >> 4;
    const bf16_t* p = base + (size_t)(tile_m + m) * stride + k0 + half * 16;
    v8bf lo = *(const v8bf*)p;
    v8bf hi = *(const v8bf*)(p + 8);
    v16bf a;
#pragma unroll
    for (int i = 0; i < 8; ++i) { a[i] = lo[i]; a[i + 8] = hi[i]; }
    return a;
}

__device__ __forceinline__ v16bf load_b_frag(const bf16_t* base, int stride,
                                             int ncol, int k0, int lane) {
    const int half = lane >> 4;
    return *(const v16bf*)(base + (size_t)ncol * stride + k0 + half * 16);
}

__device__ __forceinline__ v8f wmma_bf16(v16bf a, v16bf b, v8f c) {
    return __builtin_amdgcn_wmma_f32_16x16x32_bf16(false, a, false, b,
                                                   (short)0, c, false, false);
}

__device__ __forceinline__ float gelu_tanh(float x) {
    const float kk = 0.7978845608028654f; // sqrt(2/pi)
    float x3 = x * x * x;
    return 0.5f * x * (1.f + tanhf(kk * (x + 0.044715f * x3)));
}

// ---------------- prep kernels ----------------
__global__ __launch_bounds__(256) void zero_kernel(float* p, int n) {
    int i = blockIdx.x * 256 + threadIdx.x;
    if (i < n) p[i] = 0.f;
}

// one block per token row: l2-normalize x row (strided NCHW gather) -> bf16
__global__ __launch_bounds__(256) void xn_kernel(const float* __restrict__ x,
                                                 bf16_t* __restrict__ xn) {
    __shared__ float sred[8];
    const int n = blockIdx.x, ch = threadIdx.x;
    const int b = n >> 12, s = n & 4095;
    float v = x[(size_t)b * C_DIM * HW + (size_t)ch * HW + s];
    float ss = block_reduce_sum(v * v, sred);
    float scale = 1.f / fmaxf(sqrtf(ss), EPS_NORM);
    xn[(size_t)n * C_DIM + ch] = (bf16_t)(v * scale);
}

// one block per code: l2-normalize feat_w[:, :256] row -> bf16
__global__ __launch_bounds__(256) void mn_kernel(const float* __restrict__ fw,
                                                 bf16_t* __restrict__ mnb) {
    __shared__ float sred[8];
    const int k = blockIdx.x, ch = threadIdx.x;
    float v = fw[(size_t)k * CP + ch];
    float ss = block_reduce_sum(v * v, sred);
    float scale = 1.f / fmaxf(sqrtf(ss), EPS_NORM);
    mnb[(size_t)k * C_DIM + ch] = (bf16_t)(v * scale);
}

// transpose + pad w1 (260x256 f32) -> w1t (256 x 288 bf16)
__global__ __launch_bounds__(256) void w1t_kernel(const float* __restrict__ w1,
                                                  bf16_t* __restrict__ w1t) {
    int idx = blockIdx.x * 256 + threadIdx.x;
    if (idx >= C_DIM * P1) return;
    int j = idx / P1, kk = idx % P1;
    w1t[idx] = (bf16_t)((kk < CP) ? w1[(size_t)kk * C_DIM + j] : 0.f);
}

// transpose w2 (256x256) -> w2t (256x256 bf16)
__global__ __launch_bounds__(256) void w2t_kernel(const float* __restrict__ w2,
                                                  bf16_t* __restrict__ w2t) {
    int idx = blockIdx.x * 256 + threadIdx.x;
    int j = idx >> 8, kk = idx & 255;
    w2t[idx] = (bf16_t)w2[(size_t)kk * C_DIM + j];
}

// ---------------- GEMM 1: score = xn @ mn^T  (32768 x 2048, K=256) ----------
// block = 8 waves; wave -> 16x64 tile (4 accumulators, A reuse x4)
__global__ __launch_bounds__(256) void gemm_score(const bf16_t* __restrict__ xn,
                                                  const bf16_t* __restrict__ mnb,
                                                  float* __restrict__ score) {
    const int lane = threadIdx.x & 31, wave = threadIdx.x >> 5;
    const int tile_m = blockIdx.y * 128 + wave * 16;
    const int tile_n = blockIdx.x * 64;
    v8f acc[4] = {};
    for (int k0 = 0; k0 < C_DIM; k0 += 32) {
        v16bf a = load_a_frag(xn, C_DIM, tile_m, k0, lane);
#pragma unroll
        for (int t = 0; t < 4; ++t) {
            v16bf b = load_b_frag(mnb, C_DIM, tile_n + t * 16 + (lane & 15), k0, lane);
            acc[t] = wmma_bf16(a, b, acc[t]);
        }
    }
    const int half = lane >> 4;
#pragma unroll
    for (int t = 0; t < 4; ++t) {
        const int col = tile_n + t * 16 + (lane & 15);
#pragma unroll
        for (int r = 0; r < 8; ++r) {
            const int row = tile_m + r + half * 8;
            score[(size_t)row * K_CODE + col] = acc[t][r];
        }
    }
}

// ---- fused argmax + EMA stats + softmax (wave per row, row cached in VGPRs) -
// score row (2048 f32) is read exactly once from HBM; 64 f32 live per lane.
__global__ __launch_bounds__(256) void rowsoft_kernel(
    const float* __restrict__ score, const float* __restrict__ x,
    const float* __restrict__ y, float* __restrict__ counts,
    float* __restrict__ sums, bf16_t* __restrict__ p) {
    const int lane = threadIdx.x & 31, wave = threadIdx.x >> 5;
    const int row = blockIdx.x * 8 + wave;
    const float* sr = score + (size_t)row * K_CODE;
    float sv[64];
#pragma unroll
    for (int j = 0; j < 64; ++j) sv[j] = sr[lane + 32 * j];

    // per-lane argmax (ascending scan keeps first max), then cross-lane
    float mv = sv[0]; int mi = lane;
#pragma unroll
    for (int j = 1; j < 64; ++j)
        if (sv[j] > mv) { mv = sv[j]; mi = lane + 32 * j; }
#pragma unroll
    for (int o = 16; o; o >>= 1) {
        float ov = __shfl_xor(mv, o, 32);
        int   oi = __shfl_xor(mi, o, 32);
        if (ov > mv || (ov == mv && oi < mi)) { mv = ov; mi = oi; }
    }

    // EMA statistics
    if (lane == 0) atomicAdd(&counts[mi], 1.0f);
    const int b = row >> 12, s = row & 4095;
    for (int j = lane; j < CP; j += 32) {
        float v = (j < C_DIM)
                      ? x[(size_t)b * C_DIM * HW + (size_t)j * HW + s]
                      : y[(size_t)b * 4 * HW + (size_t)(j - C_DIM) * HW + s];
        atomicAdd(&sums[(size_t)mi * CP + j], v);
    }

    // softmax from the register-resident row
    float acc = 0.f;
#pragma unroll
    for (int j = 0; j < 64; ++j) { sv[j] = __expf(sv[j] - mv); acc += sv[j]; }
    acc = wave_reduce_sum(acc);
    const float inv = 1.f / acc;
    bf16_t* pr = p + (size_t)row * K_CODE;
#pragma unroll
    for (int j = 0; j < 64; ++j) pr[lane + 32 * j] = (bf16_t)(sv[j] * inv);
}

// ------------- EMA update + l2norm -> wnT (288 x 2048 bf16, transposed) -----
__global__ __launch_bounds__(P1) void wn_kernel(const float* __restrict__ fw,
                                                const float* __restrict__ counts,
                                                const float* __restrict__ sums,
                                                bf16_t* __restrict__ wnT) {
    __shared__ float sred[9];
    const int k = blockIdx.x, j = threadIdx.x;
    float nv = 0.f;
    if (j < CP) {
        float mean = sums[(size_t)k * CP + j] / (counts[k] + EPS_CNT);
        nv = fw[(size_t)k * CP + j] * EMA_RATE + mean * (1.f - EMA_RATE);
    }
    float ss = block_reduce_sum(nv * nv, sred);
    float scale = 1.f / fmaxf(sqrtf(ss), EPS_NORM);
    wnT[(size_t)j * K_CODE + k] = (bf16_t)(nv * scale);
}

// ------- GEMM 2: out1 = p @ wn (32768 x 288, K=2048); full N per wave -------
// 18 accumulators (144 VGPRs): each p row streamed from HBM exactly once.
__global__ __launch_bounds__(256) void gemm_attn(const bf16_t* __restrict__ p,
                                                 const bf16_t* __restrict__ wnT,
                                                 bf16_t* __restrict__ out1) {
    const int lane = threadIdx.x & 31, wave = threadIdx.x >> 5;
    const int tile_m = blockIdx.x * 128 + wave * 16;
    v8f acc[NT_P1] = {};
#pragma unroll 1
    for (int k0 = 0; k0 < K_CODE; k0 += 32) {
        v16bf a = load_a_frag(p, K_CODE, tile_m, k0, lane);
#pragma unroll
        for (int t = 0; t < NT_P1; ++t) {
            v16bf b = load_b_frag(wnT, K_CODE, t * 16 + (lane & 15), k0, lane);
            acc[t] = wmma_bf16(a, b, acc[t]);
        }
    }
    const int half = lane >> 4;
#pragma unroll
    for (int t = 0; t < NT_P1; ++t) {
        const int col = t * 16 + (lane & 15);
#pragma unroll
        for (int r = 0; r < 8; ++r)
            out1[(size_t)(tile_m + r + half * 8) * P1 + col] = (bf16_t)acc[t][r];
    }
}

// ------- GEMM 3: h = gelu(out1 @ w1 + b1) (32768 x 256, K=288) --------------
__global__ __launch_bounds__(256) void gemm_mlp1(const bf16_t* __restrict__ out1,
                                                 const bf16_t* __restrict__ w1t,
                                                 const float* __restrict__ b1,
                                                 bf16_t* __restrict__ h) {
    const int lane = threadIdx.x & 31, wave = threadIdx.x >> 5;
    const int tile_m = blockIdx.x * 128 + wave * 16;
    v8f acc[NT_C] = {};
#pragma unroll 1
    for (int k0 = 0; k0 < P1; k0 += 32) {
        v16bf a = load_a_frag(out1, P1, tile_m, k0, lane);
#pragma unroll
        for (int t = 0; t < NT_C; ++t) {
            v16bf b = load_b_frag(w1t, P1, t * 16 + (lane & 15), k0, lane);
            acc[t] = wmma_bf16(a, b, acc[t]);
        }
    }
    const int half = lane >> 4;
#pragma unroll
    for (int t = 0; t < NT_C; ++t) {
        const int col = t * 16 + (lane & 15);
        const float bias = b1[col];
#pragma unroll
        for (int r = 0; r < 8; ++r) {
            const int row = tile_m + r + half * 8;
            h[(size_t)row * C_DIM + col] = (bf16_t)gelu_tanh(acc[t][r] + bias);
        }
    }
}

// ------- GEMM 4: out = h @ w2 + b2, scatter NCHW ----------------------------
__global__ __launch_bounds__(256) void gemm_mlp2(const bf16_t* __restrict__ h,
                                                 const bf16_t* __restrict__ w2t,
                                                 const float* __restrict__ b2,
                                                 float* __restrict__ out) {
    const int lane = threadIdx.x & 31, wave = threadIdx.x >> 5;
    const int tile_m = blockIdx.x * 128 + wave * 16;
    v8f acc[NT_C] = {};
#pragma unroll 1
    for (int k0 = 0; k0 < C_DIM; k0 += 32) {
        v16bf a = load_a_frag(h, C_DIM, tile_m, k0, lane);
#pragma unroll
        for (int t = 0; t < NT_C; ++t) {
            v16bf b = load_b_frag(w2t, C_DIM, t * 16 + (lane & 15), k0, lane);
            acc[t] = wmma_bf16(a, b, acc[t]);
        }
    }
    const int half = lane >> 4;
#pragma unroll
    for (int t = 0; t < NT_C; ++t) {
        const int col = t * 16 + (lane & 15);
        const float bias = b2[col];
#pragma unroll
        for (int r = 0; r < 8; ++r) {
            const int row = tile_m + r + half * 8;
            const int b = row >> 12, s = row & 4095;
            out[(size_t)b * C_DIM * HW + (size_t)col * HW + s] = acc[t][r] + bias;
        }
    }
}

// ---------------- launch ----------------
extern "C" void kernel_launch(void* const* d_in, const int* in_sizes, int n_in,
                              void* d_out, int out_size, void* d_ws, size_t ws_size,
                              hipStream_t stream) {
    (void)in_sizes; (void)n_in; (void)out_size; (void)ws_size;
    const float* x      = (const float*)d_in[0];
    const float* y      = (const float*)d_in[1];
    const float* feat_w = (const float*)d_in[2];
    const float* w1     = (const float*)d_in[3];
    const float* b1     = (const float*)d_in[4];
    const float* w2     = (const float*)d_in[5];
    const float* b2     = (const float*)d_in[6];
    float* out = (float*)d_out;

    char* ws = (char*)d_ws;
    size_t off = 0;
    auto alloc = [&](size_t bytes) -> char* {
        char* p = ws + off;
        off += (bytes + 255) & ~(size_t)255;
        return p;
    };
    bf16_t* xn    = (bf16_t*)alloc((size_t)N_TOK * C_DIM * 2);
    bf16_t* mnb   = (bf16_t*)alloc((size_t)K_CODE * C_DIM * 2);
    float*  score = (float*)alloc((size_t)N_TOK * K_CODE * 4);
    float*  counts= (float*)alloc((size_t)K_CODE * 4);        // contiguous with
    float*  sums  = (float*)alloc((size_t)K_CODE * CP * 4);   // sums (zeroed together)
    bf16_t* p     = (bf16_t*)alloc((size_t)N_TOK * K_CODE * 2);
    bf16_t* wnT   = (bf16_t*)alloc((size_t)P1 * K_CODE * 2);
    bf16_t* out1  = (bf16_t*)alloc((size_t)N_TOK * P1 * 2);
    bf16_t* w1t   = (bf16_t*)alloc((size_t)C_DIM * P1 * 2);
    bf16_t* hbuf  = (bf16_t*)alloc((size_t)N_TOK * C_DIM * 2);
    bf16_t* w2t   = (bf16_t*)alloc((size_t)C_DIM * C_DIM * 2);

    // stats must be zero every call (deterministic relaunch)
    const int nz = K_CODE + K_CODE * CP;
    zero_kernel<<<(nz + 255) / 256, 256, 0, stream>>>(counts, nz);

    xn_kernel<<<N_TOK, 256, 0, stream>>>(x, xn);
    mn_kernel<<<K_CODE, 256, 0, stream>>>(feat_w, mnb);
    w1t_kernel<<<(C_DIM * P1 + 255) / 256, 256, 0, stream>>>(w1, w1t);
    w2t_kernel<<<(C_DIM * C_DIM) / 256, 256, 0, stream>>>(w2, w2t);

    gemm_score<<<dim3(K_CODE / 64, N_TOK / 128), 256, 0, stream>>>(xn, mnb, score);
    rowsoft_kernel<<<N_TOK / 8, 256, 0, stream>>>(score, x, y, counts, sums, p);
    wn_kernel<<<K_CODE, P1, 0, stream>>>(feat_w, counts, sums, wnT);

    gemm_attn<<<N_TOK / 128, 256, 0, stream>>>(p, wnT, out1);
    gemm_mlp1<<<N_TOK / 128, 256, 0, stream>>>(out1, w1t, b1, hbuf);
    gemm_mlp2<<<N_TOK / 128, 256, 0, stream>>>(hbuf, w2t, b2, out);
}
